// _LibXLSTM_Mixed_32770600468733
// MI455X (gfx1250) — matrix operand order repaired and checked by reading, not compiled, not run on hardware
//
#include <hip/hip_runtime.h>
#include <hip/hip_bf16.h>
#include <math.h>

typedef __attribute__((ext_vector_type(16))) _Float16 v16h;
typedef __attribute__((ext_vector_type(8)))  float    v8f;

#define BS_TOK 4096   // B*S = 64*64
#define SEQ    64
#define BATCH  64

// ---------------------------------------------------------------------------
// WMMA GEMM: C[M,N] = A[M,K] * B[K,N] (+bias[col]) (+resid[row,col])
// f32 memory, f16 WMMA inputs, f32 accumulate.
// One wave computes one 16-row M tile x two 16-col N tiles (A fragment reused
// for 2 wmma per K-step). K is a template constant -> fully unrolled loop.
// blockDim = (32,4): 4 M-tiles per block. A/B use contiguous k = half*16 + e (same order in both).
// ---------------------------------------------------------------------------
template<int K, bool HB, bool HR>
__global__ void gemm_f16_wmma(const float* __restrict__ A, int lda,
                              const float* __restrict__ Bm, int ldb,
                              float* __restrict__ C, int ldc,
                              const float* __restrict__ bias,
                              const float* __restrict__ resid, int ldr)
{
    const int tileN = blockIdx.x;                    // N/32 tiles
    const int tileM = blockIdx.y * 4 + threadIdx.y;  // M/16 tiles
    const int lane  = threadIdx.x;                   // 0..31
    const int half  = lane >> 4;
    const int l16   = lane & 15;
    const int rowA  = tileM * 16 + l16;
    const int colB0 = tileN * 32 + l16;
    const int colB1 = colB0 + 16;

    // prefetch first B rows (lowers to global_prefetch_b8)
    __builtin_prefetch(Bm + colB0, 0, 1);
    __builtin_prefetch(Bm + (size_t)ldb + colB0, 0, 1);

    v8f acc0 = {}, acc1 = {};
    #pragma unroll
    for (int k0 = 0; k0 < K; k0 += 32) {
        v16h a, b0, b1;
        // A (16x32 f16): element e -> K = half*16 + e
        #pragma unroll
        for (int e = 0; e < 16; ++e) {
            int kk = half * 16 + e;
            a[e] = (_Float16)A[(size_t)rowA * lda + k0 + kk];
        }
        // B (32x16 f16): element e -> K = half*16 + e, N = lane&15
        #pragma unroll
        for (int e = 0; e < 16; ++e) {
            int kk = half * 16 + e;
            const float* Brow = Bm + (size_t)(k0 + kk) * ldb;
            b0[e] = (_Float16)Brow[colB0];
            b1[e] = (_Float16)Brow[colB1];
        }
        acc0 = __builtin_amdgcn_wmma_f32_16x16x32_f16(
            false, a, false, b0, (short)0, acc0, false, false);
        acc1 = __builtin_amdgcn_wmma_f32_16x16x32_f16(
            false, a, false, b1, (short)0, acc1, false, false);
    }

    const float bias0 = HB ? bias[colB0] : 0.f;
    const float bias1 = HB ? bias[colB1] : 0.f;
    // C/D: element r -> row = tileM*16 + half*8 + r, col = colB
    #pragma unroll
    for (int r = 0; r < 8; ++r) {
        int row = tileM * 16 + half * 8 + r;
        float v0 = acc0[r] + bias0;
        float v1 = acc1[r] + bias1;
        if (HR) {
            v0 += resid[(size_t)row * ldr + colB0];
            v1 += resid[(size_t)row * ldr + colB1];
        }
        C[(size_t)row * ldc + colB0] = v0;
        C[(size_t)row * ldc + colB1] = v1;
    }
}

// ---------------------------------------------------------------------------
// Elementwise / helper kernels
// ---------------------------------------------------------------------------
__device__ __forceinline__ float sigmoidf_(float x) { return 1.f / (1.f + expf(-x)); }
__device__ __forceinline__ float siluf_(float x)    { return x * sigmoidf_(x); }
__device__ __forceinline__ float logsigf_(float x) {
    return (x < 0.f) ? (x - log1pf(expf(x))) : (-log1pf(expf(-x)));
}
__device__ __forceinline__ float gelu_exact_(float x) {
    return 0.5f * x * (1.f + erff(x * 0.70710678118654752f));
}

__global__ void embed_kernel(const int* __restrict__ ids,
                             const float* __restrict__ emb,
                             float* __restrict__ h)
{
    int row = blockIdx.x, t = threadIdx.x;   // blockDim 128
    h[(size_t)row * 128 + t] = emb[(size_t)ids[row] * 128 + t];
}

// LayerNorm / per-head GroupNorm (no bias, weight only). blockDim == C.
// weight index = (row % NHW)*C + t  (NHW==1 -> plain LN).
__global__ void rownorm_kernel(const float* __restrict__ x,
                               const float* __restrict__ w,
                               float* __restrict__ y, int C, int NHW)
{
    __shared__ float red[128];
    int row = blockIdx.x, t = threadIdx.x;
    float v = x[(size_t)row * C + t];
    red[t] = v; __syncthreads();
    for (int s = C >> 1; s > 0; s >>= 1) { if (t < s) red[t] += red[t + s]; __syncthreads(); }
    float mu = red[0] / (float)C; __syncthreads();
    float d = v - mu;
    red[t] = d * d; __syncthreads();
    for (int s = C >> 1; s > 0; s >>= 1) { if (t < s) red[t] += red[t + s]; __syncthreads(); }
    float var = red[0] / (float)C;
    int hix = (NHW > 1) ? (row % NHW) : 0;
    y[(size_t)row * C + t] = d * rsqrtf(var + 1e-5f) * w[hix * C + t];
}

__global__ void add_kernel(float* __restrict__ dst, const float* __restrict__ src, int n)
{
    for (int i = blockIdx.x * blockDim.x + threadIdx.x; i < n; i += gridDim.x * blockDim.x)
        dst[i] += src[i];
}

// causal depthwise conv K=4 over xm (== up[:, :256]) + SiLU -> xc
__global__ void conv_silu_kernel(const float* __restrict__ up,
                                 const float* __restrict__ cw,
                                 const float* __restrict__ cb,
                                 float* __restrict__ xc)
{
    int idx = blockIdx.x * 256 + threadIdx.x;       // BS*256 total
    int c = idx & 255, bs = idx >> 8, s = bs & 63;
    float acc = cb[c];
    #pragma unroll
    for (int k = 0; k < 4; ++k) {
        int sp = s + k - 3;
        if (sp >= 0) acc += up[(size_t)(bs - s + sp) * 512 + c] * cw[c * 4 + k];
    }
    xc[idx] = siluf_(acc);
}

// block-diagonal (LinearHeadwiseExpand) projection, block size 4
__global__ void headwise_kernel(const float* __restrict__ in, int ld,
                                const float* __restrict__ W,
                                float* __restrict__ out)
{
    int idx = blockIdx.x * 256 + threadIdx.x;       // BS*256 total
    int i = idx & 255, bs = idx >> 8;
    int nb = i >> 2, e = i & 3;
    float acc = 0.f;
    #pragma unroll
    for (int d = 0; d < 4; ++d)
        acc += in[(size_t)bs * ld + nb * 4 + d] * W[nb * 16 + d * 4 + e];
    out[idx] = acc;
}

// ig/fg gates: [BS,768] @ [768,4] + bias, 8 wave-reduced outputs per token
__global__ void gates_kernel(const float* __restrict__ q,
                             const float* __restrict__ k,
                             const float* __restrict__ v,
                             const float* __restrict__ Wig, const float* __restrict__ big,
                             const float* __restrict__ Wfg, const float* __restrict__ bfg,
                             float* __restrict__ ig, float* __restrict__ fg)
{
    int bs = blockIdx.x, t = threadIdx.x;     // blockDim 256 = 8 waves
    int o = t >> 5, lane = t & 31;
    int hh = o & 3; bool isf = (o >= 4);
    const float* W = isf ? Wfg : Wig;
    float acc = 0.f;
    for (int j = lane; j < 768; j += 32) {
        float g = (j < 256) ? q[(size_t)bs * 256 + j]
                : (j < 512) ? k[(size_t)bs * 256 + j - 256]
                            : v[(size_t)bs * 256 + j - 512];
        acc += g * W[j * 4 + hh];
    }
    for (int off = 16; off > 0; off >>= 1) acc += __shfl_down(acc, off);
    if (lane == 0) {
        float r = acc + (isf ? bfg[hh] : big[hh]);
        (isf ? fg : ig)[(size_t)bs * 4 + hh] = r;
    }
}

// stabilized parallel mLSTM attention, one block per (b, head), blockDim 64
__global__ void mlstm_attn_kernel(const float* __restrict__ q,
                                  const float* __restrict__ k,
                                  const float* __restrict__ v,
                                  const float* __restrict__ ig,
                                  const float* __restrict__ fg,
                                  float* __restrict__ hh)
{
    int b = blockIdx.x >> 2, hd = blockIdx.x & 3;
    int t = threadIdx.x;                        // 0..63 (row s)
    __shared__ float sq[64 * 64], sk[64 * 64], sv[64 * 64], slfc[64], sig[64];
    for (int d = 0; d < 64; ++d) {
        size_t src = (size_t)(b * 64 + t) * 256 + hd * 64 + d;
        sq[t * 64 + d] = q[src];
        sk[t * 64 + d] = k[src];
        sv[t * 64 + d] = v[src];
    }
    sig[t]  = ig[(size_t)(b * 64 + t) * 4 + hd];
    slfc[t] = logsigf_(fg[(size_t)(b * 64 + t) * 4 + hd]);
    __syncthreads();
    if (t == 0) {                               // inclusive prefix sum (S=64)
        float run = 0.f;
        for (int s = 0; s < 64; ++s) { run += slfc[s]; slfc[s] = run; }
    }
    __syncthreads();

    const int s = t;
    float m = -1e30f;
    for (int tt = 0; tt <= s; ++tt)
        m = fmaxf(m, slfc[s] - slfc[tt] + sig[tt]);
    float acc[64];
    #pragma unroll
    for (int d = 0; d < 64; ++d) acc[d] = 0.f;
    float csum = 0.f;
    for (int tt = 0; tt <= s; ++tt) {
        float ld = slfc[s] - slfc[tt] + sig[tt];
        float Dm = expf(ld - m);
        float qk = 0.f;
        for (int d = 0; d < 64; ++d) qk += sq[s * 64 + d] * sk[tt * 64 + d];
        float cc = qk * 0.125f * Dm;            // MDH^-0.5 = 1/8
        csum += cc;
        #pragma unroll
        for (int d = 0; d < 64; ++d) acc[d] += cc * sv[tt * 64 + d];
    }
    float norm = fmaxf(fabsf(csum), expf(-m)) + 1e-6f;
    for (int d = 0; d < 64; ++d)
        hh[(size_t)(b * 64 + s) * 256 + hd * 64 + d] = acc[d] / norm;
}

// out_inner = (head_ln(h) + skip*xc) * silu(z)
__global__ void mlstm_combine_kernel(const float* __restrict__ hn,
                                     const float* __restrict__ xc,
                                     const float* __restrict__ skip,
                                     const float* __restrict__ up,
                                     float* __restrict__ out)
{
    int idx = blockIdx.x * 256 + threadIdx.x;   // BS*256
    int c = idx & 255, bs = idx >> 8;
    float z = up[(size_t)bs * 512 + 256 + c];
    out[idx] = (hn[idx] + skip[c] * xc[idx]) * siluf_(z);
}

// gated GeLU: up2 [BS,256] -> out [BS,128]
__global__ void gelugate_kernel(const float* __restrict__ up2, float* __restrict__ out)
{
    int idx = blockIdx.x * 256 + threadIdx.x;   // BS*128
    if (idx >= BS_TOK * 128) return;
    int c = idx & 127, bs = idx >> 7;
    float a  = up2[(size_t)bs * 256 + c];
    float bb = up2[(size_t)bs * 256 + 128 + c];
    out[idx] = gelu_exact_(a) * bb;
}

// sLSTM recurrence: one block per batch row, 128 threads = (head n, chan j)
__global__ void slstm_kernel(const float* __restrict__ wx,  // [B,S,NH,4*DH] (+bias)
                             const float* __restrict__ R,   // [NH, DH, 4*DH]
                             float* __restrict__ y)          // [B,S,NH,DH]
{
    int b = blockIdx.x, t = threadIdx.x;
    int n = t >> 5, j = t & 31;
    __shared__ float hs[128];
    __shared__ float Rs[4 * 32 * 128];          // 64 KB: full R in LDS
    for (int i = t; i < 4 * 32 * 128; i += 128) Rs[i] = R[i];
    hs[t] = 0.f;
    float c = 0.f, nst = 0.f, mst = 0.f;
    __syncthreads();
    for (int s = 0; s < SEQ; ++s) {
        float ti = 0.f, tf = 0.f, tz = 0.f, to = 0.f;
        const float* Rn = Rs + n * 32 * 128;
        for (int kk = 0; kk < 32; ++kk) {
            float hk = hs[n * 32 + kk];
            const float* Rk = Rn + kk * 128;
            ti += hk * Rk[j];
            tf += hk * Rk[32 + j];
            tz += hk * Rk[64 + j];
            to += hk * Rk[96 + j];
        }
        const float* w = wx + (size_t)(b * SEQ + s) * 512 + n * 128;
        float iraw = w[j] + ti, fraw = w[32 + j] + tf;
        float zraw = w[64 + j] + tz, oraw = w[96 + j] + to;
        float lfpm = mst + logsigf_(fraw);
        float mnew = fmaxf(iraw, lfpm);
        float ii = expf(iraw - mnew);
        float ff = expf(lfpm - mnew);
        c   = ff * c + ii * tanhf(zraw);
        nst = ff * nst + ii;
        mst = mnew;
        float hnew = sigmoidf_(oraw) * c / nst;
        __syncthreads();
        hs[t] = hnew;
        y[(size_t)(b * SEQ + s) * 128 + t] = hnew;
        __syncthreads();
    }
}

// ---------------------------------------------------------------------------
// Host side
// ---------------------------------------------------------------------------
template<int K>
static inline void launch_gemm_k(const float* A, int lda, const float* B, int ldb,
                                 float* C, int ldc, int M, int N,
                                 const float* bias, const float* resid, int ldr,
                                 hipStream_t s)
{
    dim3 grid(N / 32, M / 64);
    dim3 blk(32, 4);
    if (bias) {
        if (resid) gemm_f16_wmma<K, true,  true ><<<grid, blk, 0, s>>>(A, lda, B, ldb, C, ldc, bias, resid, ldr);
        else       gemm_f16_wmma<K, true,  false><<<grid, blk, 0, s>>>(A, lda, B, ldb, C, ldc, bias, resid, ldr);
    } else {
        if (resid) gemm_f16_wmma<K, false, true ><<<grid, blk, 0, s>>>(A, lda, B, ldb, C, ldc, bias, resid, ldr);
        else       gemm_f16_wmma<K, false, false><<<grid, blk, 0, s>>>(A, lda, B, ldb, C, ldc, bias, resid, ldr);
    }
}

static inline void launch_gemm(const float* A, int lda, const float* B, int ldb,
                               float* C, int ldc, int M, int N, int K,
                               const float* bias, const float* resid, int ldr,
                               hipStream_t s)
{
    if (K == 32)       launch_gemm_k<32 >(A, lda, B, ldb, C, ldc, M, N, bias, resid, ldr, s);
    else if (K == 128) launch_gemm_k<128>(A, lda, B, ldb, C, ldc, M, N, bias, resid, ldr, s);
    else               launch_gemm_k<256>(A, lda, B, ldb, C, ldc, M, N, bias, resid, ldr, s);
}

extern "C" void kernel_launch(void* const* d_in, const int* in_sizes, int n_in,
                              void* d_out, int out_size, void* d_ws, size_t ws_size,
                              hipStream_t stream)
{
    (void)in_sizes; (void)n_in; (void)out_size; (void)ws_size;
    const int BS = BS_TOK;

    // workspace layout (floats)
    float* ws  = (float*)d_ws;
    float* h   = ws;                            // BS*128
    float* xln = h   + (size_t)BS * 128;        // BS*128
    float* up  = xln + (size_t)BS * 128;        // BS*512 (also sLSTM wx)
    float* xc  = up  + (size_t)BS * 512;        // BS*256
    float* qb  = xc  + (size_t)BS * 256;        // BS*256 (also sLSTM y)
    float* kb  = qb  + (size_t)BS * 256;        // BS*256
    float* vb  = kb  + (size_t)BS * 256;        // BS*256
    float* igb = vb  + (size_t)BS * 256;        // BS*4
    float* fgb = igb + (size_t)BS * 4;          // BS*4
    float* hhb = fgb + (size_t)BS * 4;          // BS*256
    float* t1  = hhb + (size_t)BS * 256;        // BS*256
    float* t2  = t1  + (size_t)BS * 256;        // BS*256

    auto F = [&](int i) { return (const float*)d_in[i]; };
    // pytree leaf order (sorted dict keys):
    // blocks[0..3], embed(56), final_ln_w(57), proj_W(58), proj_b(59), x(60)
    const int blkBase[4] = {0, 11, 28, 39};
    const float* embed = F(56);
    const float* fln   = F(57);
    const float* projW = F(58);
    const float* projb = F(59);
    const int*   ids   = (const int*)d_in[60];

    embed_kernel<<<BS, 128, 0, stream>>>(ids, embed, h);

    for (int blk = 0; blk < 4; ++blk) {
        int base = blkBase[blk];
        bool is_slstm = (blk % 2 == 0);
        const float *ln2, *ffup, *ffdown;

        if (is_slstm) {
            // sLSTM leaves: 0:R 1:Wf 2:Wi 3:Wo 4:Wz 5:b 6:ff_down 7:ff_up 8:gn_w 9:ln1_w 10:ln2_w
            const float* R    = F(base + 0);
            const float* bvec = F(base + 5);
            const float* gn_w = F(base + 8);
            const float* ln1  = F(base + 9);
            ln2 = F(base + 10);
            ffup = F(base + 7); ffdown = F(base + 6);
            const float* Wg[4] = { F(base + 2), F(base + 1), F(base + 4), F(base + 3) }; // i,f,z,o

            rownorm_kernel<<<BS, 128, 0, stream>>>(h, ln1, xln, 128, 1);
            // per-head, per-gate 4096x32x32 WMMA GEMMs into wx = up
            for (int n = 0; n < 4; ++n)
                for (int g = 0; g < 4; ++g)
                    launch_gemm(xln + n * 32, 128, Wg[g] + n * 1024, 32,
                                up + n * 128 + g * 32, 512, BS, 32, 32,
                                bvec + n * 128 + g * 32, nullptr, 0, stream);
            slstm_kernel<<<BATCH, 128, 0, stream>>>(up, R, qb);
            rownorm_kernel<<<BS * 4, 32, 0, stream>>>(qb, gn_w, t1, 32, 4);
            add_kernel<<<2048, 256, 0, stream>>>(h, t1, BS * 128);
        } else {
            // mLSTM leaves: 0:Wdown 1:Wfg 2:Wig 3:Wk 4:Wq 5:Wup 6:Wv 7:bfg 8:big
            //               9:conv_b 10:conv_w 11:ff_down 12:ff_up 13:ln1_w 14:ln2_w 15:mhln_w 16:skip
            const float* Wdown = F(base + 0);
            const float* Wfg   = F(base + 1);
            const float* Wig   = F(base + 2);
            const float* Wk    = F(base + 3);
            const float* Wq    = F(base + 4);
            const float* Wup   = F(base + 5);
            const float* Wv    = F(base + 6);
            const float* bfg   = F(base + 7);
            const float* big   = F(base + 8);
            const float* convb = F(base + 9);
            const float* convw = F(base + 10);
            const float* mhln  = F(base + 15);
            const float* skip  = F(base + 16);
            const float* ln1   = F(base + 13);
            ln2 = F(base + 14);
            ffup = F(base + 12); ffdown = F(base + 11);

            rownorm_kernel<<<BS, 128, 0, stream>>>(h, ln1, xln, 128, 1);
            launch_gemm(xln, 128, Wup, 512, up, 512, BS, 512, 128,
                        nullptr, nullptr, 0, stream);
            conv_silu_kernel<<<BS, 256, 0, stream>>>(up, convw, convb, xc);
            headwise_kernel<<<BS, 256, 0, stream>>>(xc, 256, Wq, qb);
            headwise_kernel<<<BS, 256, 0, stream>>>(xc, 256, Wk, kb);
            headwise_kernel<<<BS, 256, 0, stream>>>(up, 512, Wv, vb);   // v uses xm
            gates_kernel<<<BS, 256, 0, stream>>>(qb, kb, vb, Wig, big, Wfg, bfg, igb, fgb);
            mlstm_attn_kernel<<<BATCH * 4, 64, 0, stream>>>(qb, kb, vb, igb, fgb, hhb);
            rownorm_kernel<<<BS * 4, 64, 0, stream>>>(hhb, mhln, t1, 64, 4);
            mlstm_combine_kernel<<<BS, 256, 0, stream>>>(t1, xc, skip, up, t2);
            launch_gemm(t2, 256, Wdown, 128, h, 128, BS, 128, 256,
                        nullptr, h, 128, stream);
        }

        // gated-GeLU FFN (both block types)
        rownorm_kernel<<<BS, 128, 0, stream>>>(h, ln2, xln, 128, 1);
        launch_gemm(xln, 128, ffup, 256, up, 256, BS, 256, 128,
                    nullptr, nullptr, 0, stream);
        gelugate_kernel<<<2048, 256, 0, stream>>>(up, t1);
        launch_gemm(t1, 128, ffdown, 128, h, 128, BS, 128, 128,
                    nullptr, h, 128, stream);
    }

    // final LN + vocab projection (the big GEMM: 4096x128x8192)
    rownorm_kernel<<<BS, 128, 0, stream>>>(h, fln, xln, 128, 1);
    launch_gemm(xln, 128, projW, 8192, (float*)d_out, 8192, BS, 8192, 128,
                projb, nullptr, 0, stream);
}
